// MSAColGlobalAttention_18330920419588
// MI455X (gfx1250) — matrix operand order repaired and checked by reading, not compiled, not run on hardware
//
#include <hip/hip_runtime.h>
#include <hip/hip_bf16.h>
#include <math.h>

typedef __attribute__((ext_vector_type(16))) _Float16 v16h;
typedef __attribute__((ext_vector_type(8)))  float    v8f;

namespace {

constexpr int   kN = 512;
constexpr int   kL = 1024;
constexpr int   kD = 64;
constexpr float kScaleQ = 0.35355339059327373f;  // 1/sqrt(8)

__device__ __forceinline__ v8f wmma_f16(v16h a, v16h b, v8f c) {
  // D = A(16x32 f16) * B(32x16 f16) + C(16x16 f32)
  return __builtin_amdgcn_wmma_f32_16x16x32_f16(false, a, false, b, (short)0, c,
                                                false, false);
}

// A-fragment (16x32 f16) from an LDS tile x[16][stride], tile K-offset ko.
// Contiguous K order (same as load_b_frag): m=lane&15; lanes<16 hold K {0..15}+ko, lanes>=16 hold
// K {16..31}+ko.
__device__ __forceinline__ v16h load_a_frag(const _Float16* x, int stride,
                                            int ko, int lane) {
  const int m  = lane & 15;
  const int kb = (lane >> 4) * 16;
  const _Float16* row = x + m * stride;
  v16h a;
#pragma unroll
  for (int j = 0; j < 8; ++j) a[j] = row[ko + kb + j];
#pragma unroll
  for (int j = 0; j < 8; ++j) a[8 + j] = row[ko + kb + 8 + j];
  return a;
}

// B-fragment (32x16 f16) from fp32 weight W[64][outw] (row=k, col=c), col c,
// K-offset ko. Contiguous K order: n=lane&15, K = (lane>>4)*16 + j.
__device__ __forceinline__ v16h load_b_frag(const float* __restrict__ W,
                                            int outw, int c, int ko, int lane) {
  const int kb = (lane >> 4) * 16;
  v16h b;
#pragma unroll
  for (int j = 0; j < 16; ++j) b[j] = (_Float16)W[(ko + kb + j) * outw + c];
  return b;
}

// ---------------------------------------------------------------------------
// Kernel 1: LayerNorm + K/V projection (WMMA) + per-block partial sum of
// x_ln over n (for the mean-pooled query). grid(64,64), block 256 (8 waves).
// Wave w handles n = by*8+w, rows l0..l0+15. kv[row][0:8]=k, kv[row][8:16]=v.
// ---------------------------------------------------------------------------
__global__ __launch_bounds__(256) void k_ln_kv(
    const float* __restrict__ msa, const float* __restrict__ ln_g,
    const float* __restrict__ ln_b, const float* __restrict__ Wk,
    const float* __restrict__ Wv, float* __restrict__ xpart,
    float* __restrict__ kvbuf) {
  __shared__ float    xs[8][16][68];
  __shared__ _Float16 xh[8][16][72];
  const int lane = threadIdx.x & 31;
  const int w    = threadIdx.x >> 5;
  const int l0   = blockIdx.x * 16;
  const int n    = blockIdx.y * 8 + w;
  const size_t r0 = (size_t)n * kL + l0;
  const float* src = msa + r0 * kD;
  // cooperative load: 16 rows x 64 f32, float4-coalesced
#pragma unroll
  for (int it = 0; it < 8; ++it) {
    int q4 = it * 32 + lane;
    float4 v = ((const float4*)src)[q4];
    int r = q4 >> 4;
    int p = (q4 & 15) * 4;
    xs[w][r][p + 0] = v.x;
    xs[w][r][p + 1] = v.y;
    xs[w][r][p + 2] = v.z;
    xs[w][r][p + 3] = v.w;
  }
  // LayerNorm: 2 lanes per row, 32 elems each
  const int r     = lane >> 1;
  const int hbase = (lane & 1) * 32;
  float s = 0.f, sq = 0.f;
#pragma unroll
  for (int e = 0; e < 32; ++e) {
    float x = xs[w][r][hbase + e];
    s  += x;
    sq += x * x;
  }
  s  += __shfl_xor(s, 1, 32);
  sq += __shfl_xor(sq, 1, 32);
  const float mu  = s * (1.f / 64.f);
  const float var = sq * (1.f / 64.f) - mu * mu;
  const float rsd = rsqrtf(var + 1e-5f);
#pragma unroll
  for (int e = 0; e < 32; ++e) {
    int   cc = hbase + e;
    float x  = (xs[w][r][cc] - mu) * rsd;
    x = x * ln_g[cc] + ln_b[cc];
    xs[w][r][cc] = x;                 // f32 copy for the xbar partial sum
    xh[w][r][cc] = (_Float16)x;       // f16 copy for WMMA
  }
  // K/V: X(16x64) @ [Wk|Wv](64x16) with two K=32 WMMA steps
  v16h a0 = load_a_frag(&xh[w][0][0], 72, 0, lane);
  v16h a1 = load_a_frag(&xh[w][0][0], 72, 32, lane);
  const int c  = lane & 15;
  const int kb = (lane >> 4) * 16;
  const float* Wcol = (c < 8) ? (Wk + c) : (Wv + (c - 8));  // pointer select
  v16h b0, b1;
#pragma unroll
  for (int j = 0; j < 16; ++j) {
    b0[j] = (_Float16)Wcol[(kb + j) * 8];
    b1[j] = (_Float16)Wcol[(32 + kb + j) * 8];
  }
  v8f acc = {};
  acc = wmma_f16(a0, b0, acc);
  acc = wmma_f16(a1, b1, acc);
  // branch-free, coalesced store: per slot i, lanes 0-15 write one 64B row,
  // lanes 16-31 write another.
  const int mb = (lane >> 4) * 8;
  float* dst = kvbuf + (r0 + (size_t)mb) * 16 + c;
#pragma unroll
  for (int i = 0; i < 8; ++i) dst[i * 16] = acc[i];
  __syncthreads();
  // deterministic partial sum over this block's 8 sequences
  for (int t = threadIdx.x; t < 16 * 64; t += 256) {
    int rr = t >> 6, e = t & 63;
    float sum = 0.f;
#pragma unroll
    for (int ww = 0; ww < 8; ++ww) sum += xs[ww][rr][e];
    xpart[(size_t)blockIdx.y * (kL * kD) + (size_t)(l0 + rr) * kD + e] = sum;
  }
}

// ---------------------------------------------------------------------------
// Kernel 1b: fixed-order reduction of the 64 partials -> xbar[L*64]
// ---------------------------------------------------------------------------
__global__ __launch_bounds__(256) void k_reduce_xbar(
    const float* __restrict__ xpart, float* __restrict__ xbar) {
  int i = blockIdx.x * 256 + threadIdx.x;  // 0..65535
  float s = 0.f;
  for (int by = 0; by < 64; ++by) s += xpart[(size_t)by * (kL * kD) + i];
  xbar[i] = s;
}

// ---------------------------------------------------------------------------
// Kernel 2: q = (xbar/N)*Wq*SCALE, attention over n, softmax, PV.
// grid 512, block 64 (2 waves); wave w handles l = bx*2+w.
// ---------------------------------------------------------------------------
__global__ __launch_bounds__(64) void k_attn(
    const float* __restrict__ xbar, const float* __restrict__ Wq,
    const float* __restrict__ kvbuf, float* __restrict__ gout) {
  __shared__ float sc[2][8][512];
  __shared__ float qrow[2][64];
  __shared__ float xbr[2][64];
  __shared__ float seL[2][8];
  const int lane = threadIdx.x & 31;
  const int w    = threadIdx.x >> 5;
  const int l    = blockIdx.x * 2 + w;
  const float fac = kScaleQ / (float)kN;
  xbr[w][lane]      = xbar[l * 64 + lane] * fac;
  xbr[w][lane + 32] = xbar[l * 64 + lane + 32] * fac;
  float q0 = 0.f, q1 = 0.f;
  for (int k = 0; k < 64; ++k) {
    float xv = xbr[w][k];
    q0 += xv * Wq[k * 64 + lane];
    q1 += xv * Wq[k * 64 + lane + 32];
  }
  qrow[w][lane]      = q0;
  qrow[w][lane + 32] = q1;
  // pass 1: scores + per-head max
  float mh[8];
#pragma unroll
  for (int h = 0; h < 8; ++h) mh[h] = -3.0e38f;
  for (int j = 0; j < 16; ++j) {
    int nn = j * 32 + lane;
    const float* kp = kvbuf + ((size_t)nn * kL + l) * 16;  // k in cols 0..7
    float k8[8];
#pragma unroll
    for (int d = 0; d < 8; ++d) k8[d] = kp[d];
#pragma unroll
    for (int h = 0; h < 8; ++h) {
      float sv = 0.f;
#pragma unroll
      for (int d = 0; d < 8; ++d) sv += qrow[w][h * 8 + d] * k8[d];
      sc[w][h][nn] = sv;
      mh[h] = fmaxf(mh[h], sv);
    }
  }
#pragma unroll
  for (int h = 0; h < 8; ++h)
#pragma unroll
    for (int off = 16; off > 0; off >>= 1)
      mh[h] = fmaxf(mh[h], __shfl_xor(mh[h], off, 32));
  // pass 2: exp + per-head sum
  float se[8];
#pragma unroll
  for (int h = 0; h < 8; ++h) se[h] = 0.f;
  for (int j = 0; j < 16; ++j) {
    int nn = j * 32 + lane;
#pragma unroll
    for (int h = 0; h < 8; ++h) {
      float e = __expf(sc[w][h][nn] - mh[h]);
      sc[w][h][nn] = e;
      se[h] += e;
    }
  }
#pragma unroll
  for (int h = 0; h < 8; ++h) {
#pragma unroll
    for (int off = 16; off > 0; off >>= 1) se[h] += __shfl_xor(se[h], off, 32);
    if (lane == h) seL[w][h] = se[h];
  }
  // pass 3: out[h,d] = sum_n p * v ; lane -> (h,d)=lane and lane+32
  const int h0 = lane >> 3,        d0 = lane & 7;
  const int h1 = (lane + 32) >> 3, d1 = lane & 7;
  float o0 = 0.f, o1 = 0.f;
  for (int nn = 0; nn < kN; ++nn) {
    const float* vp = kvbuf + ((size_t)nn * kL + l) * 16 + 8;  // v in cols 8..15
    o0 += sc[w][h0][nn] * vp[d0];
    o1 += sc[w][h1][nn] * vp[d1];
  }
  gout[l * 64 + lane]      = o0 / seL[w][h0];
  gout[l * 64 + lane + 32] = o1 / seL[w][h1];
}

// ---------------------------------------------------------------------------
// Kernel 3: re-LN, gate = sigmoid(X*Wg + bg), T = gate*gout[l], Y = T*Wo + bo.
// grid(64,64), block 256 (8 waves); wave: 16 rows of one n. 16 WMMAs/wave.
// ---------------------------------------------------------------------------
__global__ __launch_bounds__(256) void k_gate_out(
    const float* __restrict__ msa, const float* __restrict__ ln_g,
    const float* __restrict__ ln_b, const float* __restrict__ Wg,
    const float* __restrict__ bg, const float* __restrict__ Wo,
    const float* __restrict__ bo, const float* __restrict__ gout,
    float* __restrict__ out) {
  __shared__ float    xs[8][16][68];
  __shared__ _Float16 xh[8][16][72];
  const int lane = threadIdx.x & 31;
  const int w    = threadIdx.x >> 5;
  const int l0   = blockIdx.x * 16;
  const int n    = blockIdx.y * 8 + w;
  const size_t r0 = (size_t)n * kL + l0;
  const float* src = msa + r0 * kD;
#pragma unroll
  for (int it = 0; it < 8; ++it) {
    int q4 = it * 32 + lane;
    float4 v = ((const float4*)src)[q4];
    int r = q4 >> 4;
    int p = (q4 & 15) * 4;
    xs[w][r][p + 0] = v.x;
    xs[w][r][p + 1] = v.y;
    xs[w][r][p + 2] = v.z;
    xs[w][r][p + 3] = v.w;
  }
  const int r     = lane >> 1;
  const int hbase = (lane & 1) * 32;
  float s = 0.f, sq = 0.f;
#pragma unroll
  for (int e = 0; e < 32; ++e) {
    float x = xs[w][r][hbase + e];
    s  += x;
    sq += x * x;
  }
  s  += __shfl_xor(s, 1, 32);
  sq += __shfl_xor(sq, 1, 32);
  const float mu  = s * (1.f / 64.f);
  const float var = sq * (1.f / 64.f) - mu * mu;
  const float rsd = rsqrtf(var + 1e-5f);
#pragma unroll
  for (int e = 0; e < 32; ++e) {
    int   cc = hbase + e;
    float x  = (xs[w][r][cc] - mu) * rsd;
    xh[w][r][cc] = (_Float16)(x * ln_g[cc] + ln_b[cc]);
  }
  v16h a0 = load_a_frag(&xh[w][0][0], 72, 0, lane);
  v16h a1 = load_a_frag(&xh[w][0][0], 72, 32, lane);
  // reuse this wave's (now-dead) xs slice as the f16 gated tile
  _Float16* th = reinterpret_cast<_Float16*>(&xs[w][0][0]);
  const int cl = lane & 15;
  const int mb = (lane >> 4) * 8;
#pragma unroll
  for (int nt = 0; nt < 4; ++nt) {
    const int c = nt * 16 + cl;
    v16h b0 = load_b_frag(Wg, 64, c, 0, lane);
    v16h b1 = load_b_frag(Wg, 64, c, 32, lane);
    v8f acc;
    const float bias = bg[c];
#pragma unroll
    for (int i = 0; i < 8; ++i) acc[i] = bias;   // fold +bg into accumulator
    acc = wmma_f16(a0, b0, acc);
    acc = wmma_f16(a1, b1, acc);
#pragma unroll
    for (int i = 0; i < 8; ++i) {
      int   m = mb + i;
      float g = 1.f / (1.f + __expf(-acc[i]));   // sigmoid gate
      float t = g * gout[(l0 + m) * 64 + c];     // gate * broadcast attn out
      th[m * 72 + c] = (_Float16)t;
    }
  }
  v16h c0 = load_a_frag(th, 72, 0, lane);
  v16h c1 = load_a_frag(th, 72, 32, lane);
#pragma unroll
  for (int nt = 0; nt < 4; ++nt) {
    const int c = nt * 16 + cl;
    v16h b0 = load_b_frag(Wo, 64, c, 0, lane);
    v16h b1 = load_b_frag(Wo, 64, c, 32, lane);
    v8f acc;
    const float bias = bo[c];
#pragma unroll
    for (int i = 0; i < 8; ++i) acc[i] = bias;   // fold +bo into accumulator
    acc = wmma_f16(c0, b0, acc);
    acc = wmma_f16(c1, b1, acc);
#pragma unroll
    for (int i = 0; i < 8; ++i)
      out[(r0 + (size_t)(mb + i)) * kD + c] = acc[i];
  }
}

}  // namespace

extern "C" void kernel_launch(void* const* d_in, const int* in_sizes, int n_in,
                              void* d_out, int out_size, void* d_ws,
                              size_t ws_size, hipStream_t stream) {
  (void)in_sizes; (void)n_in; (void)out_size; (void)ws_size;
  const float* msa  = (const float*)d_in[0];
  const float* ln_g = (const float*)d_in[1];
  const float* ln_b = (const float*)d_in[2];
  const float* Wq   = (const float*)d_in[3];
  const float* Wk   = (const float*)d_in[4];
  const float* Wv   = (const float*)d_in[5];
  const float* Wg   = (const float*)d_in[6];
  const float* bg   = (const float*)d_in[7];
  const float* Wo   = (const float*)d_in[8];
  const float* bo   = (const float*)d_in[9];
  float* out = (float*)d_out;

  char* ws = (char*)d_ws;
  const size_t xpart_bytes = (size_t)64 * kL * kD * sizeof(float);   // 16 MB
  const size_t xbar_bytes  = (size_t)kL * kD * sizeof(float);        // 256 KB
  const size_t kv_bytes    = (size_t)kN * kL * 16 * sizeof(float);   // 32 MB
  float* xpart = (float*)ws;
  float* xbar  = (float*)(ws + xpart_bytes);
  float* kvbuf = (float*)(ws + xpart_bytes + xbar_bytes);
  float* gout  = (float*)(ws + xpart_bytes + xbar_bytes + kv_bytes);

  dim3 gridA(kL / 16, kN / 8);  // (64, 64)
  k_ln_kv<<<gridA, 256, 0, stream>>>(msa, ln_g, ln_b, Wk, Wv, xpart, kvbuf);
  k_reduce_xbar<<<(kL * kD) / 256, 256, 0, stream>>>(xpart, xbar);
  k_attn<<<kL / 2, 64, 0, stream>>>(xbar, Wq, kvbuf, gout);
  k_gate_out<<<gridA, 256, 0, stream>>>(msa, ln_g, ln_b, Wg, bg, Wo, bo, gout,
                                        out);
}
